// Model_27273042330265
// MI455X (gfx1250) — compile-verified, matrix operand order checked by reading, not run on hardware
//
#include <hip/hip_runtime.h>
#include <math.h>
#include <stdint.h>

typedef _Float16 h16;
typedef __attribute__((ext_vector_type(8)))  _Float16 v8h;
typedef __attribute__((ext_vector_type(16))) _Float16 v16h;
typedef __attribute__((ext_vector_type(8)))  float    v8f;
typedef int i4v __attribute__((vector_size(16)));   // matches builtin's V4i

#define BB      16
#define SEQL    512
#define ENC_IN  321
#define PRED    96
#define DM      512
#define DSTATE  16
#define DCONV   4
#define DI      1024
#define DTRANK  32
#define LTOK    ENC_IN
#define ROWS    (BB*LTOK)     // 5136 token rows
#define EPSF    1e-5f

// ---- WMMA GEMM tiling ----
#define BM 128
#define BN 64
#define BK 32

#if defined(__HIP_DEVICE_COMPILE__) && __has_builtin(__builtin_amdgcn_global_load_async_to_lds_b128)
#define USE_ASYNC 1
#else
#define USE_ASYNC 0
#endif

// ---------------------------------------------------------------- async helpers
template <int N>
__device__ inline void wait_async() {
#if __has_builtin(__builtin_amdgcn_s_wait_asynccnt)
  __builtin_amdgcn_s_wait_asynccnt(N);
#else
  asm volatile("s_wait_asynccnt %0" ::"i"(N));
#endif
}

__device__ inline void async_ld_b128(const void* gsrc, void* ldst) {
#if USE_ASYNC
  __builtin_amdgcn_global_load_async_to_lds_b128(
      (__attribute__((address_space(1))) i4v*)(unsigned long long)(uintptr_t)gsrc,
      (__attribute__((address_space(3))) i4v*)(unsigned)(uintptr_t)ldst,
      0, 0);
#else
  (void)gsrc; (void)ldst;
#endif
}

// ---------------------------------------------------------------- utilities
__global__ __launch_bounds__(256) void f32_to_f16_kernel(const float* __restrict__ s,
                                                         h16* __restrict__ d, int n) {
  int i = blockIdx.x * 256 + threadIdx.x;
  if (i < n) d[i] = (h16)s[i];
}

// RevIN stats + normalize + transpose: xt16[(b*321+c)*512 + s] (f16 GEMM input)
__global__ __launch_bounds__(256) void revin_kernel(const float* __restrict__ x,
    const float* __restrict__ rw, const float* __restrict__ rb,
    float* __restrict__ meanO, float* __restrict__ stdO, h16* __restrict__ xt16) {
  int warp = threadIdx.x >> 5, lane = threadIdx.x & 31;
  int g = blockIdx.x * 8 + warp;
  if (g >= ROWS) return;
  int b = g / ENC_IN, c = g % ENC_IN;
  const float* px = x + (size_t)b * SEQL * ENC_IN + c;
  float s = 0.f, ss = 0.f;
  for (int t = lane; t < SEQL; t += 32) {
    float v = px[(size_t)t * ENC_IN];
    s += v; ss += v * v;
  }
  for (int o = 16; o > 0; o >>= 1) { s += __shfl_xor(s, o); ss += __shfl_xor(ss, o); }
  float mean = s * (1.f / SEQL);
  float var  = ss * (1.f / SEQL) - mean * mean;
  float sd   = sqrtf(var + EPSF);
  if (lane == 0) { meanO[g] = mean; stdO[g] = sd; }
  float w = rw[c], bv = rb[c], inv = 1.f / sd;
  h16* pd = xt16 + (size_t)g * SEQL;
  for (int t = lane; t < SEQL; t += 32) {
    float v = px[(size_t)t * ENC_IN];
    pd[t] = (h16)((v - mean) * inv * w + bv);
  }
}

// ---------------------------------------------------------------- WMMA GEMM
// C(MxN) = A(MxK, row-major, stride lda) * W(NxK, row-major)^T  [+bias][softplus]
__device__ inline v16h frag_from_lds(const h16* p, int hi) {
  v8h lo  = *(const v8h*)(p + hi * 8);
  v8h hiv = *(const v8h*)(p + 16 + hi * 8);
  return __builtin_shufflevector(lo, hiv, 0,1,2,3,4,5,6,7,8,9,10,11,12,13,14,15);
}

__device__ inline void store_tile(v8f acc, int row0, int col, int M, int N,
                                  const float* bias, float* Cf, h16* Ch, int mode) {
  if (col >= N) return;
  float bv = bias ? bias[col] : 0.f;
#pragma unroll
  for (int v = 0; v < 8; ++v) {
    int row = row0 + v;
    if (row >= M) continue;
    float val = acc[v] + bv;
    if (mode & 1) val = (val > 20.f) ? val : log1pf(__expf(val));  // softplus
    size_t o = (size_t)row * N + col;
    if (Cf) Cf[o] = val;
    if (Ch) Ch[o] = (h16)val;
  }
}

__global__ __launch_bounds__(256) void gemm_wmma_kernel(
    const h16* __restrict__ A, int lda,
    const h16* __restrict__ W,
    const float* __restrict__ bias,
    float* __restrict__ Cf, h16* __restrict__ Ch,
    int M, int N, int K, int mode) {
  __shared__ __align__(16) h16 sA[2][BM][BK];
  __shared__ __align__(16) h16 sB[2][BN][BK];

  int tid  = threadIdx.x;
  int wave = tid >> 5, lane = tid & 31;
  int r = lane & 15, hi = lane >> 4;
  int m0 = blockIdx.y * BM;
  int n0 = blockIdx.x * BN;

  v8f acc0 = {}, acc1 = {}, acc2 = {}, acc3 = {};

  int ar = tid >> 1, ak = (tid & 1) * 16;   // A stage: 2 threads/row, 16 halves each
  int br = tid >> 2, bk = (tid & 3) * 8;    // B stage: 4 threads/row, 8 halves each
  // Clamp OOB rows/cols into range: clamped data only feeds never-stored outputs.
  int arow = m0 + ar; if (arow >= M) arow = M - 1;
  int brow = n0 + br; if (brow >= N) brow = N - 1;

  auto compute = [&](int buf) {
    v16h af  = frag_from_lds(&sA[buf][wave * 16 + r][0], hi);
    v16h bf0 = frag_from_lds(&sB[buf][ 0 + r][0], hi);
    v16h bf1 = frag_from_lds(&sB[buf][16 + r][0], hi);
    v16h bf2 = frag_from_lds(&sB[buf][32 + r][0], hi);
    v16h bf3 = frag_from_lds(&sB[buf][48 + r][0], hi);
    acc0 = __builtin_amdgcn_wmma_f32_16x16x32_f16(false, af, false, bf0, (short)0, acc0, false, false);
    acc1 = __builtin_amdgcn_wmma_f32_16x16x32_f16(false, af, false, bf1, (short)0, acc1, false, false);
    acc2 = __builtin_amdgcn_wmma_f32_16x16x32_f16(false, af, false, bf2, (short)0, acc2, false, false);
    acc3 = __builtin_amdgcn_wmma_f32_16x16x32_f16(false, af, false, bf3, (short)0, acc3, false, false);
  };

#if USE_ASYNC
  // ---- double-buffered LDS with ASYNCcnt-tracked direct-to-LDS loads ----
  auto prefetch = [&](int buf, int k0) {
    const h16* ga = A + (size_t)arow * lda + k0 + ak;
    async_ld_b128(ga,     &sA[buf][ar][ak]);
    async_ld_b128(ga + 8, &sA[buf][ar][ak + 8]);
    const h16* gb = W + (size_t)brow * K + k0 + bk;
    async_ld_b128(gb, &sB[buf][br][bk]);
  };

  int nk = K / BK;
  prefetch(0, 0);
  for (int i = 0; i < nk; ++i) {
    int cur = i & 1;
    if (i + 1 < nk) {
      prefetch(cur ^ 1, (i + 1) * BK);
      wait_async<3>();      // only the 3 next-stage loads may remain in flight
    } else {
      wait_async<0>();
    }
    __syncthreads();        // current stage resident for all waves
    compute(cur);
    __syncthreads();        // done reading before buffer is overwritten
  }
#else
  // ---- fallback: synchronous staging through VGPRs ----
  for (int k0 = 0; k0 < K; k0 += BK) {
    const h16* pA = A + (size_t)arow * lda + k0 + ak;
    v8h a0 = *(const v8h*)pA;
    v8h a1 = *(const v8h*)(pA + 8);
    v8h b0 = *(const v8h*)(W + (size_t)brow * K + k0 + bk);
    *(v8h*)&sA[0][ar][ak]     = a0;
    *(v8h*)&sA[0][ar][ak + 8] = a1;
    *(v8h*)&sB[0][br][bk]     = b0;
    __syncthreads();
    compute(0);
    __syncthreads();
  }
#endif

  int mrow = m0 + wave * 16 + 8 * hi;   // C layout: lane half selects row group
  store_tile(acc0, mrow, n0 +  0 + r, M, N, bias, Cf, Ch, mode);
  store_tile(acc1, mrow, n0 + 16 + r, M, N, bias, Cf, Ch, mode);
  store_tile(acc2, mrow, n0 + 32 + r, M, N, bias, Cf, Ch, mode);
  store_tile(acc3, mrow, n0 + 48 + r, M, N, bias, Cf, Ch, mode);
}

// ------------------------------------------------- causal depthwise conv + SiLU
__global__ __launch_bounds__(256) void conv_silu_kernel(
    const h16* __restrict__ xz,            // rows x 2048, xr = cols [0,1024)
    const float* __restrict__ cw, const float* __restrict__ cb,
    h16* __restrict__ xc) {
  size_t idx = (size_t)blockIdx.x * 256 + threadIdx.x;
  if (idx >= (size_t)ROWS * DI) return;
  int d   = (int)(idx % DI);
  int row = (int)(idx / DI);
  int l   = row % LTOK;
  int row0 = row - l;
  float w0 = cw[d*4+0], w1 = cw[d*4+1], w2 = cw[d*4+2], w3 = cw[d*4+3];
  float acc = cb[d];
  const h16* base = xz + (size_t)row0 * (2*DI) + d;
  if (l >= 3) acc += w0 * (float)base[(size_t)(l-3) * (2*DI)];
  if (l >= 2) acc += w1 * (float)base[(size_t)(l-2) * (2*DI)];
  if (l >= 1) acc += w2 * (float)base[(size_t)(l-1) * (2*DI)];
  acc += w3 * (float)base[(size_t)l * (2*DI)];
  float s = acc / (1.f + __expf(-acc));   // SiLU
  xc[idx] = (h16)s;
}

// ------------------------------------------------- selective scan (+D*x, z-gate)
__global__ __launch_bounds__(256) void scan_kernel(
    const h16* __restrict__ delta, const h16* __restrict__ xc,
    const h16* __restrict__ zbase,          // xz + DI (row stride 2*DI)
    const float* __restrict__ dbl,          // rows x 64; B at 32, C at 48
    const float* __restrict__ Alog, const float* __restrict__ Dp,
    h16* __restrict__ y16) {
  int d = blockIdx.x * 256 + threadIdx.x;   // 0..DI-1
  int b = blockIdx.y;
  float a[DSTATE], h[DSTATE];
#pragma unroll
  for (int n = 0; n < DSTATE; ++n) { a[n] = -__expf(Alog[d * DSTATE + n]); h[n] = 0.f; }
  float Dd = Dp[d];
  __shared__ float sB[DSTATE], sC[DSTATE];
  for (int l = 0; l < LTOK; ++l) {
    size_t row = (size_t)b * LTOK + l;
    if (threadIdx.x < 32) {
      float v = dbl[row * 64 + 32 + threadIdx.x];
      if (threadIdx.x < 16) sB[threadIdx.x] = v; else sC[threadIdx.x - 16] = v;
    }
    __syncthreads();
    float dt = (float)delta[row * DI + d];
    float xt = (float)xc[row * DI + d];
    float zt = (float)zbase[row * (2*DI) + d];
    float bx = dt * xt;
    float y = 0.f;
#pragma unroll
    for (int n = 0; n < DSTATE; ++n) {
      float dA = __expf(dt * a[n]);
      h[n] = dA * h[n] + bx * sB[n];
      y += h[n] * sC[n];
    }
    y += Dd * xt;
    float sz = zt / (1.f + __expf(-zt));
    y16[row * DI + d] = (h16)(y * sz);
    __syncthreads();
  }
}

// ------------------------------------------------- flip / combine / tail
__global__ __launch_bounds__(256) void reverse16_kernel(const h16* __restrict__ s,
                                                        h16* __restrict__ d) {
  size_t idx = (size_t)blockIdx.x * 256 + threadIdx.x;
  if (idx >= (size_t)ROWS * DM) return;
  int col = (int)(idx % DM);
  int row = (int)(idx / DM);
  int l = row % LTOK, b = row / LTOK;
  size_t rr = (size_t)b * LTOK + (LTOK - 1 - l);
  d[rr * DM + col] = s[idx];
}

__global__ __launch_bounds__(256) void combine_kernel(
    const float* __restrict__ fwd, const float* __restrict__ bwd,
    float* __restrict__ t, h16* __restrict__ t16, h16* __restrict__ trev) {
  size_t idx = (size_t)blockIdx.x * 256 + threadIdx.x;
  if (idx >= (size_t)ROWS * DM) return;
  int col = (int)(idx % DM);
  int row = (int)(idx / DM);
  int l = row % LTOK, b = row / LTOK;
  size_t rr = (size_t)b * LTOK + (LTOK - 1 - l);
  float v = fwd[idx] + bwd[rr * DM + col] + t[idx];
  t[idx] = v;
  t16[idx] = (h16)v;
  trev[rr * DM + col] = (h16)v;
}

__global__ __launch_bounds__(256) void final_kernel(
    const float* __restrict__ head, const float* __restrict__ rw,
    const float* __restrict__ rb, const float* __restrict__ mean,
    const float* __restrict__ sd, float* __restrict__ out) {
  int idx = blockIdx.x * 256 + threadIdx.x;
  if (idx >= BB * PRED * ENC_IN) return;
  int c = idx % ENC_IN;
  int p = (idx / ENC_IN) % PRED;
  int b = idx / (ENC_IN * PRED);
  int g = b * ENC_IN + c;
  float v = head[(size_t)g * PRED + p];
  v = (v - rb[c]) / (rw[c] + EPSF * EPSF);
  out[idx] = v * sd[g] + mean[g];
}

// ---------------------------------------------------------------- launch
static inline void launch_gemm(const h16* A, int lda, const h16* W, const float* bias,
                               float* Cf, h16* Ch, int M, int N, int K, int mode,
                               hipStream_t s) {
  dim3 g((N + BN - 1) / BN, (M + BM - 1) / BM);
  gemm_wmma_kernel<<<g, dim3(256), 0, s>>>(A, lda, W, bias, Cf, Ch, M, N, K, mode);
}

extern "C" void kernel_launch(void* const* d_in, const int* in_sizes, int n_in,
                              void* d_out, int out_size, void* d_ws, size_t ws_size,
                              hipStream_t stream) {
  (void)in_sizes; (void)n_in; (void)out_size; (void)ws_size;
  const float* x     = (const float*)d_in[0];
  const float* rw    = (const float*)d_in[1];
  const float* rb    = (const float*)d_in[2];
  const float* lin1w = (const float*)d_in[3];
  const float* lin1b = (const float*)d_in[4];
  const float* headw = (const float*)d_in[5];
  const float* headb = (const float*)d_in[6];
  const float* ipw   = (const float*)d_in[7];
  const float* cw    = (const float*)d_in[8];
  const float* cb    = (const float*)d_in[9];
  const float* xpw   = (const float*)d_in[10];
  const float* dtw   = (const float*)d_in[11];
  const float* dtb   = (const float*)d_in[12];
  const float* alog  = (const float*)d_in[13];
  const float* Dp    = (const float*)d_in[14];
  const float* opw   = (const float*)d_in[15];
  float* out = (float*)d_out;

  size_t off = 0;
  auto alloc = [&](size_t bytes) -> void* {
    off = (off + 255) & ~(size_t)255;
    void* p = (char*)d_ws + off;
    off += bytes;
    return p;
  };

  h16*   xt16  = (h16*)  alloc((size_t)ROWS * SEQL * 2);
  float* mean  = (float*)alloc((size_t)ROWS * 4);
  float* sdv   = (float*)alloc((size_t)ROWS * 4);
  float* tf    = (float*)alloc((size_t)ROWS * DM * 4);
  h16*   t16   = (h16*)  alloc((size_t)ROWS * DM * 2);
  h16*   trev  = (h16*)  alloc((size_t)ROWS * DM * 2);
  float* fwd   = (float*)alloc((size_t)ROWS * DM * 4);
  float* bwd   = (float*)alloc((size_t)ROWS * DM * 4);
  float* headO = (float*)alloc((size_t)ROWS * PRED * 4);
  h16*   lin16 = (h16*)  alloc((size_t)DM * SEQL * 2);
  h16*   hd16  = (h16*)  alloc((size_t)PRED * DM * 2);
  h16*   ip16  = (h16*)  alloc((size_t)2 * 2 * DI * DM * 2);
  h16*   xp16  = (h16*)  alloc((size_t)2 * 64 * DI * 2);
  h16*   dtp16 = (h16*)  alloc((size_t)2 * DI * DTRANK * 2);
  h16*   op16  = (h16*)  alloc((size_t)2 * DM * DI * 2);
  h16*   xz16  = (h16*)  alloc((size_t)ROWS * 2 * DI * 2);
  h16*   xc16  = (h16*)  alloc((size_t)ROWS * DI * 2);
  float* dbl32 = (float*)alloc((size_t)ROWS * 64 * 4);
  h16*   dbl16 = (h16*)  alloc((size_t)ROWS * 64 * 2);
  h16*   del16 = (h16*)  alloc((size_t)ROWS * DI * 2);
  h16*   y16   = (h16*)  alloc((size_t)ROWS * DI * 2);

  auto cvt = [&](const float* s, h16* d, int n) {
    f32_to_f16_kernel<<<dim3((n + 255) / 256), dim3(256), 0, stream>>>(s, d, n);
  };
  cvt(lin1w, lin16, DM * SEQL);
  cvt(headw, hd16, PRED * DM);
  cvt(ipw,   ip16, 2 * 2 * DI * DM);
  cvt(xpw,   xp16, 2 * 64 * DI);
  cvt(dtw,   dtp16, 2 * DI * DTRANK);
  cvt(opw,   op16, 2 * DM * DI);

  revin_kernel<<<dim3((ROWS + 7) / 8), dim3(256), 0, stream>>>(x, rw, rb, mean, sdv, xt16);

  // embed: t = xn^T @ lin1_w^T + b  (f32 + f16 copies)
  launch_gemm(xt16, SEQL, lin16, lin1b, tf, t16, ROWS, DM, SEQL, 0, stream);
  reverse16_kernel<<<dim3((int)(((size_t)ROWS * DM + 255) / 256)), dim3(256), 0, stream>>>(t16, trev);

  for (int layer = 0; layer < 2; ++layer) {
    for (int dir = 0; dir < 2; ++dir) {
      const h16* Ain = dir ? trev : t16;
      launch_gemm(Ain, DM, ip16 + (size_t)dir * 2 * DI * DM, nullptr,
                  nullptr, xz16, ROWS, 2 * DI, DM, 0, stream);
      conv_silu_kernel<<<dim3((int)(((size_t)ROWS * DI + 255) / 256)), dim3(256), 0, stream>>>(
          xz16, cw + (size_t)dir * DI * DCONV, cb + (size_t)dir * DI, xc16);
      launch_gemm(xc16, DI, xp16 + (size_t)dir * 64 * DI, nullptr,
                  dbl32, dbl16, ROWS, 64, DI, 0, stream);
      launch_gemm(dbl16, 64, dtp16 + (size_t)dir * DI * DTRANK, dtb + (size_t)dir * DI,
                  nullptr, del16, ROWS, DI, DTRANK, 1, stream);   // softplus fused
      scan_kernel<<<dim3(DI / 256, BB), dim3(256), 0, stream>>>(
          del16, xc16, xz16 + DI, dbl32,
          alog + (size_t)dir * DI * DSTATE, Dp + (size_t)dir * DI, y16);
      launch_gemm(y16, DI, op16 + (size_t)dir * DM * DI, nullptr,
                  dir ? bwd : fwd, nullptr, ROWS, DM, DI, 0, stream);
    }
    combine_kernel<<<dim3((int)(((size_t)ROWS * DM + 255) / 256)), dim3(256), 0, stream>>>(
        fwd, bwd, tf, t16, trev);
  }

  launch_gemm(t16, DM, hd16, headb, headO, nullptr, ROWS, PRED, DM, 0, stream);
  final_kernel<<<dim3((BB * PRED * ENC_IN + 255) / 256), dim3(256), 0, stream>>>(
      headO, rw, rb, mean, sdv, out);
}
